// AnomalyTransformer_43138651521286
// MI455X (gfx1250) — matrix operand order repaired and checked by reading, not compiled, not run on hardware
//
#include <hip/hip_runtime.h>

// ---------------------------------------------------------------------------
// Anomaly Transformer forward on gfx1250 (MI455X), wave32 + WMMA f16.
// B=4, L=512, CIN=38, D=512, H=8, NL=3, DFF=2048, E=64
// Bandwidth-bound by series/prior outputs (201 MB); GEMMs in f16 WMMA.
// ---------------------------------------------------------------------------

typedef __attribute__((ext_vector_type(16))) _Float16 v16h;
typedef __attribute__((ext_vector_type(8)))  float    v8f;

static constexpr int Bb  = 4;
static constexpr int Ls  = 512;
static constexpr int CIn = 38;
static constexpr int Dd  = 512;
static constexpr int Hh  = 8;
static constexpr int NLs = 3;
static constexpr int DF  = 2048;
static constexpr int Ee  = 64;
static constexpr int BL  = Bb * Ls;   // 2048 rows
static constexpr int BH  = Bb * Hh;   // 32 attention batches

// GEMM tiling
static constexpr int GBM = 64;
static constexpr int GBN = 128;
static constexpr int GBK = 32;
static constexpr int LDS_A = GBK + 8;   // padded half-stride
static constexpr int LDS_B = GBK + 8;

// ---------------------------------------------------------------------------
// f32 -> f16 transpose-convert: in (s,K,N) -> out (s,N,K)
// ---------------------------------------------------------------------------
__global__ __launch_bounds__(256) void f32_to_f16_t_kernel(
    const float* __restrict__ in, _Float16* __restrict__ out,
    int K, int N, long total) {
  long i = (long)blockIdx.x * 256 + threadIdx.x;
  long stride = (long)gridDim.x * 256;
  long KN = (long)K * N;
  for (; i < total; i += stride) {
    long s = i / KN;
    long r = i - s * KN;
    int n = (int)(r / K);
    int k = (int)(r - (long)n * K);
    out[i] = (_Float16)in[(s * K + k) * N + n];
  }
}

// ---------------------------------------------------------------------------
// Token embedding: circular conv1d(k=3) + sinusoidal positional embedding
// ---------------------------------------------------------------------------
__global__ __launch_bounds__(256) void embed_kernel(
    const float* __restrict__ x, const float* __restrict__ tokw,
    float* __restrict__ enc32, _Float16* __restrict__ enc16) {
  int bl = blockIdx.x;            // 0 .. B*L-1
  int b = bl / Ls, l = bl % Ls;
  __shared__ float xs[3 * CIn];
  int tid = threadIdx.x;
  if (tid < 3 * CIn) {
    int w = tid / CIn, i = tid % CIn;
    int lw = (l + w - 1 + Ls) % Ls;  // wrap pad
    xs[tid] = x[((long)b * Ls + lw) * CIn + i];
  }
  __syncthreads();
  for (int d = tid; d < Dd; d += 256) {
    float acc = 0.f;
#pragma unroll 6
    for (int j = 0; j < 3 * CIn; ++j) acc += xs[j] * tokw[(long)j * Dd + d];
    int i2 = d & ~1;
    float div = expf(-(float)i2 * (logf(10000.f) / (float)Dd));
    float ang = (float)l * div;
    float pe = (d & 1) ? cosf(ang) : sinf(ang);
    float v = acc + pe;
    enc32[(long)bl * Dd + d] = v;
    enc16[(long)bl * Dd + d] = (_Float16)v;
  }
}

// ---------------------------------------------------------------------------
// Generic batched WMMA GEMM: C = act(alpha * A(f16,MxK) @ B^T + bias)
//   B is ALWAYS n-major: element (k,n) read from Bm[n*ldb + k].
//   ACT: 0 = none, 1 = exact GELU
// Block: 256 threads (8 waves). Block tile 64x128, wave tile 32x32.
// K, lda, ldb multiples of 32; M multiple of 64. Only N is guarded.
// A tile staged via global_load_async_to_lds_b128 (ASYNCcnt path).
// ---------------------------------------------------------------------------
template <int ACT>
__global__ __launch_bounds__(256) void gemm_wmma(
    int M, int N, int K, float alpha,
    const _Float16* __restrict__ A, int lda, long sA1, long sA2,
    const _Float16* __restrict__ Bm, int ldb, long sB1, long sB2,
    const float* __restrict__ bias,
    float* __restrict__ C32, int ldc, long sC1, long sC2,
    _Float16* __restrict__ C16,
    _Float16* __restrict__ C16T,   // optional (m/Ls, n, m%Ls) f16 store (for V^T)
    int Z2) {
  int z = blockIdx.z;
  int z1 = z / Z2, z2 = z % Z2;
  A  += z1 * sA1 + z2 * sA2;
  Bm += z1 * sB1 + z2 * sB2;
  long coff = z1 * sC1 + z2 * sC2;

  __shared__ _Float16 As[GBM * LDS_A];
  __shared__ _Float16 Bt[GBN * LDS_B];   // transposed: Bt[n][k]

  const int tid  = threadIdx.x;
  const int lane = tid & 31;
  const int wave = tid >> 5;
  const int wm = wave >> 2;   // 0..1
  const int wn = wave & 3;    // 0..3
  const int m0 = blockIdx.y * GBM;
  const int n0 = blockIdx.x * GBN;

  // per-thread staging coordinates
  const int a_row = tid >> 2;            // 64 rows, 4 threads/row
  const int a_c8  = (tid & 3) * 8;       // 8 halves (16B) per thread
  const unsigned a_lds = (unsigned)(size_t)&As[a_row * LDS_A + a_c8];
  const int b_n  = tid >> 1;             // 128 rows, 2 threads/row
  const int b_kh = (tid & 1) * 16;       // 16 halves per thread

  v8f acc[2][2] = {};

  for (int k0 = 0; k0 < K; k0 += GBK) {
    __syncthreads();
    // ---- A tile: one async b128 Memory->LDS per thread ----
    {
      unsigned long long gsrc =
          (unsigned long long)(A + (long)(m0 + a_row) * lda + k0 + a_c8);
      asm volatile("global_load_async_to_lds_b128 %0, %1, off"
                   :: "v"(a_lds), "v"(gsrc) : "memory");
      if (k0 + GBK < K)  // prefetch next B tile stream
        __builtin_prefetch(Bm + (long)(n0 + b_n) * ldb + k0 + GBK, 0, 1);
    }
    // ---- B tile (n-major) -> Bt[n][k], 2x uint4 per thread ----
    {
      int gn = n0 + b_n;
      uint4 v0 = make_uint4(0, 0, 0, 0), v1 = v0;
      if (gn < N) {
        const uint4* Bg =
            (const uint4*)(Bm + (long)gn * ldb + k0 + b_kh);
        v0 = Bg[0];
        v1 = Bg[1];
      }
      uint4* dst = (uint4*)&Bt[b_n * LDS_B + b_kh];
      dst[0] = v0;
      dst[1] = v1;
    }
    asm volatile("s_wait_asynccnt 0x0" ::: "memory");
    __syncthreads();

    // ---- fragments per ISA 7.12.2 layouts ----
    union F16Frag { v16h v; unsigned int u[8]; };
    F16Frag afrag[2], bfrag[2];
    const int klo = (lane < 16) ? 0 : 8;
    const int arow = lane & 15;
#pragma unroll
    for (int tm = 0; tm < 2; ++tm) {
      int r = wm * 32 + tm * 16 + arow;
      const unsigned int* src = (const unsigned int*)&As[r * LDS_A];
#pragma unroll
      for (int i = 0; i < 4; ++i) afrag[tm].u[i]     = src[(klo >> 1) + i];
#pragma unroll
      for (int i = 0; i < 4; ++i) afrag[tm].u[4 + i] = src[((16 + klo) >> 1) + i];
    }
    const int kb = (lane < 16) ? 0 : 16;
#pragma unroll
    for (int tn = 0; tn < 2; ++tn) {
      int n = wn * 32 + tn * 16 + (lane & 15);
      const unsigned int* src = (const unsigned int*)&Bt[n * LDS_B];
#pragma unroll
      for (int i = 0; i < 8; ++i) bfrag[tn].u[i] = src[(kb >> 2) + i + (i & 4)];
    }
#pragma unroll
    for (int tm = 0; tm < 2; ++tm)
#pragma unroll
      for (int tn = 0; tn < 2; ++tn)
        acc[tm][tn] = __builtin_amdgcn_wmma_f32_16x16x32_f16(
            false, afrag[tm].v, false, bfrag[tn].v,
            (short)0, acc[tm][tn], false, false);
  }

  // ---- epilogue ----
  float* c32 = C32 ? C32 + coff : nullptr;
  _Float16* c16 = C16 ? C16 + coff : nullptr;
  const int moff = (lane >> 4) * 8;
  const int ncol = lane & 15;
#pragma unroll
  for (int tm = 0; tm < 2; ++tm)
#pragma unroll
    for (int tn = 0; tn < 2; ++tn)
#pragma unroll
      for (int r = 0; r < 8; ++r) {
        int m = m0 + wm * 32 + tm * 16 + r + moff;
        int n = n0 + wn * 32 + tn * 16 + ncol;
        if (n < N) {
          float v = acc[tm][tn][r] * alpha;
          if (bias) v += bias[n];
          if (ACT == 1) v = 0.5f * v * (1.0f + erff(v * 0.70710678118f));
          if (c32) c32[(long)m * ldc + n] = v;
          if (c16) c16[(long)m * ldc + n] = (_Float16)v;
          if (C16T)  // transposed (b, n, l) layout for V^T
            C16T[((long)(m >> 9) * Dd + n) * Ls + (m & 511)] = (_Float16)v;
        }
      }
}

// ---------------------------------------------------------------------------
// Fused scores + softmax + series writeback:
//   one block = 32 rows x full 512 cols of softmax(scale * Q K^T) for one (b,h)
// Eliminates the 33.5 MB f32 scores round-trip entirely.
// ---------------------------------------------------------------------------
__global__ __launch_bounds__(256) void scores_softmax_kernel(
    const _Float16* __restrict__ q, const _Float16* __restrict__ kk,
    float* __restrict__ ser32, _Float16* __restrict__ ser16) {
  const int z = blockIdx.z;            // b*H + h
  const int b = z / Hh, h = z % Hh;
  const _Float16* qg = q  + ((long)b * Ls) * Dd + h * Ee;
  const _Float16* kg = kk + ((long)b * Ls) * Dd + h * Ee;
  float* o32 = ser32 + (long)z * Ls * Ls;
  _Float16* o16 = ser16 + (long)z * Ls * Ls;
  const int m0 = blockIdx.y * 32;

  __shared__ _Float16 Aq[32 * LDS_A];
  __shared__ _Float16 Bt[Ls * LDS_B];          // all 512 K-rows, 32 halves each
  __shared__ float red[2][16][4];

  const int tid  = threadIdx.x;
  const int lane = tid & 31;
  const int wave = tid >> 5;
  const int wm = wave >> 2;   // 0..1   (16-row half)
  const int wn = wave & 3;    // 0..3   (128-col strip)

  v8f acc[8] = {};

  for (int k0 = 0; k0 < Ee; k0 += GBK) {       // 2 iterations
    __syncthreads();
    if (tid < 128) {                           // A: 32x32 halves, 1 uint4/thread
      int row = tid >> 2, c8 = (tid & 3) * 8;
      *(uint4*)&Aq[row * LDS_A + c8] =
          *(const uint4*)(qg + (long)(m0 + row) * Dd + k0 + c8);
    }
#pragma unroll
    for (int rr = 0; rr < 2; ++rr) {           // B: 512x32 halves, 8 uint4/thread
      int n = tid * 2 + rr;
      const uint4* src = (const uint4*)(kg + (long)n * Dd + k0);
      uint4* dst = (uint4*)&Bt[n * LDS_B];
#pragma unroll
      for (int j = 0; j < 4; ++j) dst[j] = src[j];
    }
    __syncthreads();

    union F16Frag { v16h v; unsigned int u[8]; };
    F16Frag af, bf;
    const int klo = (lane < 16) ? 0 : 8;
    {
      const unsigned int* src =
          (const unsigned int*)&Aq[(wm * 16 + (lane & 15)) * LDS_A];
#pragma unroll
      for (int i = 0; i < 4; ++i) af.u[i]     = src[(klo >> 1) + i];
#pragma unroll
      for (int i = 0; i < 4; ++i) af.u[4 + i] = src[((16 + klo) >> 1) + i];
    }
    const int kb = (lane < 16) ? 0 : 16;
#pragma unroll
    for (int tn = 0; tn < 8; ++tn) {
      int n = wn * 128 + tn * 16 + (lane & 15);
      const unsigned int* src = (const unsigned int*)&Bt[n * LDS_B];
#pragma unroll
      for (int i = 0; i < 8; ++i) bf.u[i] = src[(kb >> 2) + i + (i & 4)];
      acc[tn] = __builtin_amdgcn_wmma_f32_16x16x32_f16(
          false, af.v, false, bf.v, (short)0, acc[tn], false, false);
    }
  }

  // ---- softmax over full rows (scale, max, exp, sum, normalize) ----
  const float scale = 0.125f;  // 1/sqrt(E)
  const int half = lane >> 4;  // 0: rows r, 1: rows r+8
  float pmax[8];
#pragma unroll
  for (int r = 0; r < 8; ++r) {
    float m = -3.4e38f;
#pragma unroll
    for (int tn = 0; tn < 8; ++tn) {
      acc[tn][r] *= scale;
      m = fmaxf(m, acc[tn][r]);
    }
    for (int off = 1; off < 16; off <<= 1) m = fmaxf(m, __shfl_xor(m, off, 32));
    pmax[r] = m;
  }
  if ((lane & 15) == 0)
#pragma unroll
    for (int r = 0; r < 8; ++r) red[wm][r + half * 8][wn] = pmax[r];
  __syncthreads();
  float rmax[8];
#pragma unroll
  for (int r = 0; r < 8; ++r) {
    float m = red[wm][r + half * 8][0];
#pragma unroll
    for (int w = 1; w < 4; ++w) m = fmaxf(m, red[wm][r + half * 8][w]);
    rmax[r] = m;
  }
  __syncthreads();
  float psum[8];
#pragma unroll
  for (int r = 0; r < 8; ++r) {
    float s = 0.f;
#pragma unroll
    for (int tn = 0; tn < 8; ++tn) {
      acc[tn][r] = expf(acc[tn][r] - rmax[r]);
      s += acc[tn][r];
    }
    for (int off = 1; off < 16; off <<= 1) s += __shfl_xor(s, off, 32);
    psum[r] = s;
  }
  if ((lane & 15) == 0)
#pragma unroll
    for (int r = 0; r < 8; ++r) red[wm][r + half * 8][wn] = psum[r];
  __syncthreads();
#pragma unroll
  for (int r = 0; r < 8; ++r) {
    float s = red[wm][r + half * 8][0] + red[wm][r + half * 8][1] +
              red[wm][r + half * 8][2] + red[wm][r + half * 8][3];
    float inv = 1.f / s;
    long row = m0 + wm * 16 + r + half * 8;
#pragma unroll
    for (int tn = 0; tn < 8; ++tn) {
      int col = wn * 128 + tn * 16 + (lane & 15);
      float v = acc[tn][r] * inv;
      o32[row * Ls + col] = v;
      o16[row * Ls + col] = (_Float16)v;
    }
  }
}

// ---------------------------------------------------------------------------
// (optional residual) + LayerNorm; writes f32 and/or f16
// ---------------------------------------------------------------------------
__global__ __launch_bounds__(256) void add_layernorm(
    const float* __restrict__ a, const float* __restrict__ rb,
    const float* __restrict__ g, const float* __restrict__ bb,
    float* __restrict__ o32, _Float16* __restrict__ o16) {
  long row = blockIdx.x;
  const float* pa = a + row * Dd;
  const float* pb = rb ? rb + row * Dd : nullptr;
  __shared__ float red[16];
  int tid = threadIdx.x;
  float s = 0.f, s2 = 0.f;
  float vals[2];
  int cnt = 0;
  for (int i = tid; i < Dd; i += 256) {
    float v = pa[i] + (pb ? pb[i] : 0.f);
    vals[cnt++] = v;
    s += v; s2 += v * v;
  }
  for (int off = 16; off; off >>= 1) {
    s += __shfl_xor(s, off, 32);
    s2 += __shfl_xor(s2, off, 32);
  }
  if ((tid & 31) == 0) { red[tid >> 5] = s; red[8 + (tid >> 5)] = s2; }
  __syncthreads();
  float ts = 0.f, ts2 = 0.f;
#pragma unroll
  for (int w = 0; w < 8; ++w) { ts += red[w]; ts2 += red[8 + w]; }
  float mean = ts / Dd;
  float var = ts2 / Dd - mean * mean;
  float inv = rsqrtf(var + 1e-5f);
  cnt = 0;
  for (int i = tid; i < Dd; i += 256) {
    float v = (vals[cnt++] - mean) * inv * g[i] + bb[i];
    if (o32) o32[row * Dd + i] = v;
    if (o16) o16[row * Dd + i] = (_Float16)v;
  }
}

// ---------------------------------------------------------------------------
// Gaussian prior from sigma head: one row (b,h,l) per block
// ---------------------------------------------------------------------------
__global__ __launch_bounds__(256) void sigma_prior(
    const float* __restrict__ sig /* (B,L,H) */, float* __restrict__ pout) {
  long idx = blockIdx.x;   // b*H*L + h*L + l
  int l = (int)(idx % Ls);
  int h = (int)((idx / Ls) % Hh);
  long b = idx / ((long)Ls * Hh);
  float sraw = sig[(b * Ls + l) * Hh + h];
  float sg = 1.f / (1.f + expf(-5.f * sraw));
  float sigma = powf(3.f, sg + 1e-5f) - 1.f;
  float inv2 = 1.f / (2.f * sigma * sigma);
  float coef = 0.3989422804014327f / sigma;   // 1/(sqrt(2*pi)*sigma)
  float* orow = pout + idx * Ls;
  for (int s = threadIdx.x; s < Ls; s += 256) {
    float d = (float)(l - s);
    orow[s] = expf(-d * d * inv2) * coef;
  }
}

// ---------------------------------------------------------------------------
// Host orchestration
// ---------------------------------------------------------------------------
extern "C" void kernel_launch(void* const* d_in, const int* in_sizes, int n_in,
                              void* d_out, int out_size, void* d_ws, size_t ws_size,
                              hipStream_t stream) {
  const float* x     = (const float*)d_in[0];
  const float* tokw  = (const float*)d_in[1];
  const float* Wq    = (const float*)d_in[2];
  const float* bq    = (const float*)d_in[3];
  const float* Wk    = (const float*)d_in[4];
  const float* bk    = (const float*)d_in[5];
  const float* Wv    = (const float*)d_in[6];
  const float* bv    = (const float*)d_in[7];
  const float* Wsg   = (const float*)d_in[8];
  const float* bsg   = (const float*)d_in[9];
  const float* Wo    = (const float*)d_in[10];
  const float* bo    = (const float*)d_in[11];
  const float* c1w   = (const float*)d_in[12];
  const float* c1b   = (const float*)d_in[13];
  const float* c2w   = (const float*)d_in[14];
  const float* c2b   = (const float*)d_in[15];
  const float* ln1g  = (const float*)d_in[16];
  const float* ln1b  = (const float*)d_in[17];
  const float* ln2g  = (const float*)d_in[18];
  const float* ln2b  = (const float*)d_in[19];
  const float* lnfg  = (const float*)d_in[20];
  const float* lnfb  = (const float*)d_in[21];
  const float* prjw  = (const float*)d_in[22];
  const float* prjb  = (const float*)d_in[23];

  // output layout: out (B,L,CIN) | series (NL,B,H,L,L) | prior (NL,B,H,L,L)
  float* out_f      = (float*)d_out;
  float* series_out = out_f + (long)Bb * Ls * CIn;
  float* prior_out  = series_out + (long)NLs * BH * Ls * Ls;

  // workspace bump allocator (~72 MB total)
  char* wp = (char*)d_ws;
  auto alloc = [&](size_t bytes) -> void* {
    void* p = (void*)wp;
    wp += (bytes + 255) & ~(size_t)255;
    return p;
  };
  float*    encf   = (float*)   alloc((size_t)BL * Dd * 4);
  _Float16* enc16  = (_Float16*)alloc((size_t)BL * Dd * 2);
  float*    x1f    = (float*)   alloc((size_t)BL * Dd * 4);
  _Float16* x116   = (_Float16*)alloc((size_t)BL * Dd * 2);
  _Float16* q16    = (_Float16*)alloc((size_t)BL * Dd * 2);
  _Float16* k16    = (_Float16*)alloc((size_t)BL * Dd * 2);
  _Float16* vT16   = (_Float16*)alloc((size_t)BL * Dd * 2);   // (B, D, L)
  _Float16* attn16 = (_Float16*)alloc((size_t)BL * Dd * 2);
  float*    projf  = (float*)   alloc((size_t)BL * Dd * 4);   // reused for FFN out
  float*    sigf   = (float*)   alloc((size_t)BL * Hh * 4);
  _Float16* ser16  = (_Float16*)alloc((size_t)BH * Ls * Ls * 2);
  _Float16* y16    = (_Float16*)alloc((size_t)BL * DF * 2);
  // transposed f16 weights (n-major)
  _Float16* WqT = (_Float16*)alloc((size_t)NLs * Dd * Dd * 2);
  _Float16* WkT = (_Float16*)alloc((size_t)NLs * Dd * Dd * 2);
  _Float16* WvT = (_Float16*)alloc((size_t)NLs * Dd * Dd * 2);
  _Float16* WoT = (_Float16*)alloc((size_t)NLs * Dd * Dd * 2);
  _Float16* WsT = (_Float16*)alloc((size_t)NLs * Dd * Hh * 2);
  _Float16* c1T = (_Float16*)alloc((size_t)NLs * Dd * DF * 2);
  _Float16* c2T = (_Float16*)alloc((size_t)NLs * DF * Dd * 2);
  _Float16* prT = (_Float16*)alloc((size_t)Dd * CIn * 2);

  auto cvtT = [&](const float* src, _Float16* dst, int K, int N, int slabs) {
    long total = (long)slabs * K * N;
    int blocks = (int)((total + 255) / 256);
    if (blocks > 4096) blocks = 4096;
    f32_to_f16_t_kernel<<<blocks, 256, 0, stream>>>(src, dst, K, N, total);
  };
  cvtT(Wq, WqT, Dd, Dd, NLs);
  cvtT(Wk, WkT, Dd, Dd, NLs);
  cvtT(Wv, WvT, Dd, Dd, NLs);
  cvtT(Wo, WoT, Dd, Dd, NLs);
  cvtT(Wsg, WsT, Dd, Hh, NLs);
  cvtT(c1w, c1T, Dd, DF, NLs);
  cvtT(c2w, c2T, DF, Dd, NLs);
  cvtT(prjw, prT, Dd, CIn, 1);

  // token embedding + positional encoding
  embed_kernel<<<BL, 256, 0, stream>>>(x, tokw, encf, enc16);

  for (int l = 0; l < NLs; ++l) {
    const _Float16* wq = WqT + (long)l * Dd * Dd;
    const _Float16* wk = WkT + (long)l * Dd * Dd;
    const _Float16* wv = WvT + (long)l * Dd * Dd;
    const _Float16* wo = WoT + (long)l * Dd * Dd;
    const _Float16* ws = WsT + (long)l * Dd * Hh;
    const _Float16* w1 = c1T + (long)l * Dd * DF;
    const _Float16* w2 = c2T + (long)l * DF * Dd;
    float* series_l = series_out + (long)l * BH * Ls * Ls;
    float* prior_l  = prior_out  + (long)l * BH * Ls * Ls;

    // Q/K projections (f16), V projection stored transposed (B,D,L)
    gemm_wmma<0><<<dim3(Dd / GBN, BL / GBM, 1), 256, 0, stream>>>(
        BL, Dd, Dd, 1.f, enc16, Dd, 0, 0, wq, Dd, 0, 0,
        bq + (long)l * Dd, nullptr, Dd, 0, 0, q16, nullptr, 1);
    gemm_wmma<0><<<dim3(Dd / GBN, BL / GBM, 1), 256, 0, stream>>>(
        BL, Dd, Dd, 1.f, enc16, Dd, 0, 0, wk, Dd, 0, 0,
        bk + (long)l * Dd, nullptr, Dd, 0, 0, k16, nullptr, 1);
    gemm_wmma<0><<<dim3(Dd / GBN, BL / GBM, 1), 256, 0, stream>>>(
        BL, Dd, Dd, 1.f, enc16, Dd, 0, 0, wv, Dd, 0, 0,
        bv + (long)l * Dd, nullptr, Dd, 0, 0, nullptr, vT16, 1);

    // sigma head (B,L,H) f32 + Gaussian prior
    gemm_wmma<0><<<dim3(1, BL / GBM, 1), 256, 0, stream>>>(
        BL, Hh, Dd, 1.f, enc16, Dd, 0, 0, ws, Dd, 0, 0,
        bsg + (long)l * Hh, sigf, Hh, 0, 0, nullptr, nullptr, 1);
    sigma_prior<<<BH * Ls, 256, 0, stream>>>(sigf, prior_l);

    // fused: softmax(scale * Q K^T) -> series (f32 -> d_out, f16 -> ws)
    scores_softmax_kernel<<<dim3(1, Ls / 32, BH), 256, 0, stream>>>(
        q16, k16, series_l, ser16);

    // attn_out = series @ V, batched over (b,h); B = V^T (n-major)
    gemm_wmma<0><<<dim3(1, Ls / GBM, BH), 256, 0, stream>>>(
        Ls, Ee, Ls, 1.f,
        ser16, Ls, (long)Hh * Ls * Ls, (long)Ls * Ls,
        vT16, Ls, (long)Dd * Ls, (long)Ee * Ls,
        nullptr, nullptr, Dd, (long)Ls * Dd, (long)Ee, attn16, nullptr, Hh);

    // output projection (f32 for residual)
    gemm_wmma<0><<<dim3(Dd / GBN, BL / GBM, 1), 256, 0, stream>>>(
        BL, Dd, Dd, 1.f, attn16, Dd, 0, 0, wo, Dd, 0, 0,
        bo + (long)l * Dd, projf, Dd, 0, 0, nullptr, nullptr, 1);

    // x1 = LN(enc + attn_proj)
    add_layernorm<<<BL, 256, 0, stream>>>(
        encf, projf, ln1g + (long)l * Dd, ln1b + (long)l * Dd, x1f, x116);

    // FFN: GELU(x1 @ W1 + b1) @ W2 + b2
    gemm_wmma<1><<<dim3(DF / GBN, BL / GBM, 1), 256, 0, stream>>>(
        BL, DF, Dd, 1.f, x116, Dd, 0, 0, w1, Dd, 0, 0,
        c1b + (long)l * DF, nullptr, DF, 0, 0, y16, nullptr, 1);
    gemm_wmma<0><<<dim3(Dd / GBN, BL / GBM, 1), 256, 0, stream>>>(
        BL, Dd, DF, 1.f, y16, DF, 0, 0, w2, DF, 0, 0,
        c2b + (long)l * Dd, projf, Dd, 0, 0, nullptr, nullptr, 1);

    // enc = LN(x1 + ffn)
    add_layernorm<<<BL, 256, 0, stream>>>(
        x1f, projf, ln2g + (long)l * Dd, ln2b + (long)l * Dd, encf, enc16);
  }

  // final LN (f16 only) + projection to CIN
  add_layernorm<<<BL, 256, 0, stream>>>(encf, nullptr, lnfg, lnfb, nullptr, x116);
  gemm_wmma<0><<<dim3(1, BL / GBM, 1), 256, 0, stream>>>(
      BL, CIn, Dd, 1.f, x116, Dd, 0, 0, prT, Dd, 0, 0,
      prjb, out_f, CIn, 0, 0, nullptr, nullptr, 1);
}
